// SparseAttention_53326313947855
// MI455X (gfx1250) — matrix operand order repaired and checked by reading, not compiled, not run on hardware
//
#include <hip/hip_runtime.h>

// ---------------------------------------------------------------------------
// Block-sparse attention for MI455X (gfx1250), bf16 WMMA path with
// async global->LDS weight staging.
//   B=1, S=4096, H=1024, NH=16, HD=64, BS=64
// ---------------------------------------------------------------------------

typedef __bf16 bf16;
typedef bf16 v16bf __attribute__((ext_vector_type(16)));
typedef bf16 v8bf  __attribute__((ext_vector_type(8)));
typedef float v8f  __attribute__((ext_vector_type(8)));
typedef int   v4i  __attribute__((ext_vector_type(4)));

#define S_  4096
#define H_  1024
#define NH_ 16
#define HD_ 64
#define BS_ 64
#define NB_ (S_ / BS_)   // 64 blocks per side

#ifndef __has_builtin
#define __has_builtin(x) 0
#endif

#if __has_builtin(__builtin_amdgcn_global_load_async_to_lds_b128)
#define HAVE_ASYNC_LDS 1
#else
#define HAVE_ASYNC_LDS 0
#endif

#if HAVE_ASYNC_LDS
#if __has_builtin(__builtin_amdgcn_s_wait_asynccnt)
#define WAIT_ASYNC(n) __builtin_amdgcn_s_wait_asynccnt((short)(n))
#else
#define WAIT_ASYNC(n) asm volatile("s_wait_asynccnt %0" ::"i"(n))
#endif
// pointer types per clang's diagnostic: param0 = v4i addrspace(1)*, param1 = LDS
typedef __attribute__((address_space(1))) v4i* glb_v4i_p;
typedef __attribute__((address_space(3))) v4i* lds_v4i_p;
#else
#define WAIT_ASYNC(n)
#endif

// One 16-byte chunk per lane: global -> LDS.  Async (ASYNCcnt) when available,
// else a synchronous register bounce (compiler handles loadcnt/dscnt waits).
__device__ __forceinline__ void stage16(const bf16* __restrict__ gsrc, bf16* ldst) {
#if HAVE_ASYNC_LDS
  __builtin_amdgcn_global_load_async_to_lds_b128(
      (glb_v4i_p)(uintptr_t)gsrc,
      (lds_v4i_p)(unsigned)(uintptr_t)ldst,
      0, 0);
#else
  *(v8bf*)ldst = *(const v8bf*)gsrc;
#endif
}

// ---- WMMA fragment loaders (per ISA 7.12.2 tables) -------------------------

// A-matrix 16x32 bf16: lane m = lane&15, half = lane>>4.
// per lane: bf16[0..7]  = A[m][half*8 + 0..7]
//           bf16[8..15] = A[m][16 + half*8 + 0..7]
__device__ __forceinline__ v16bf load_fragA(const bf16* base, int ld) {
  int lane = threadIdx.x & 31;
  int m = lane & 15, hf = lane >> 4;
  const bf16* p = base + (size_t)m * ld + hf * 8;
  union { v16bf v; v8bf h[2]; } f;
  f.h[0] = *(const v8bf*)(p);
  f.h[1] = *(const v8bf*)(p + 16);
  return f.v;
}

// B-matrix 32x16 bf16 (column-major source given as row-major Wrow[n][k]):
// lane n = lane&15, khalf = lane>>4; bf16[0..7] = B[k0 + khalf*8 + 0..7][n],
// bf16[8..15] = B[k0 + 16 + khalf*8 + 0..7][n]
__device__ __forceinline__ v16bf load_fragB(const bf16* base, int ld) {
  int lane = threadIdx.x & 31;
  int n = lane & 15, kh = lane >> 4;
  const bf16* p = base + (size_t)n * ld + kh * 8;
  union { v16bf v; v8bf h[2]; } f;
  f.h[0] = *(const v8bf*)(p);
  f.h[1] = *(const v8bf*)(p + 16);
  return f.v;
}

__device__ __forceinline__ v8f wmma_bf16(v16bf a, v16bf b, v8f c) {
  // (neg_a, A, neg_b, B, c_mod, C, reuse_a, reuse_b)
  return __builtin_amdgcn_wmma_f32_16x16x32_bf16(false, a, false, b,
                                                 (short)0, c, false, false);
}

// ---- fp32 -> bf16 convert ---------------------------------------------------
__global__ void cvt_f32_bf16(const float* __restrict__ x, bf16* __restrict__ y, int n) {
  for (int i = blockIdx.x * blockDim.x + threadIdx.x; i < n;
       i += gridDim.x * blockDim.x)
    y[i] = (bf16)x[i];
}

// ---------------------------------------------------------------------------
// Shared GEMM core: 256 threads = 8 waves; workgroup computes 128(m) x 64(n).
// Weight tile 64(n) x 32(k) bf16 is async-staged into LDS (double buffered,
// pitch 40 elements = 20-bank stride -> conflict-free ds_load_b128 fragments).
// ---------------------------------------------------------------------------
#define BPITCH 40                       // LDS row pitch in bf16 elements
#define BTILE  (64 * BPITCH)            // one buffer: 5120 B

// ---- fused QKV projection: [4096,1024] x [1024,1024]^T (x3) ----------------
// blockIdx.z: 0->Q, 1->K, 2->V.  Q/K stored [head][s][64] bf16, V stored
// transposed [head][64][s] bf16 so P@V B-fragments are contiguous.
__global__ void __launch_bounds__(256)
qkv_gemm(const bf16* __restrict__ X,
         const bf16* __restrict__ Wq, const bf16* __restrict__ Wk,
         const bf16* __restrict__ Wv,
         const float* __restrict__ bq, const float* __restrict__ bk,
         const float* __restrict__ bv,
         bf16* __restrict__ Q, bf16* __restrict__ K, bf16* __restrict__ Vt) {
  __shared__ bf16 bstage[2][BTILE];
  const int z = blockIdx.z;
  const bf16*  W    = (z == 0) ? Wq : (z == 1) ? Wk : Wv;
  const float* bias = (z == 0) ? bq : (z == 1) ? bk : bv;
  const int wave = threadIdx.x >> 5, lane = threadIdx.x & 31;
  const int n0 = blockIdx.x * 64;
  const int m0 = blockIdx.y * 128 + wave * 16;
  const int n = lane & 15, hf = lane >> 4;

  // staging map: thread i copies 16 B: row = i/4 (of 64), chunk = i%4 (of 4)
  const int crow = threadIdx.x >> 2, cchunk = threadIdx.x & 3;
  const bf16* gsrc = W + (size_t)(n0 + crow) * H_ + cchunk * 8;
  bf16* ldst = &bstage[0][crow * BPITCH + cchunk * 8];

  v8f c[4];
  for (int t = 0; t < 4; t++) {
    float b = bias[n0 + t * 16 + n];            // bias is per output column
    for (int r = 0; r < 8; r++) c[t][r] = b;
  }

  stage16(gsrc, ldst);                          // tile 0 -> buffer 0
  for (int s = 0; s < H_ / 32; s++) {
    const int k0 = s * 32;
    // issue next tile (last step re-issues tile 31 into the idle buffer)
    const int knx = (s + 1 < H_ / 32) ? (k0 + 32) : k0;
    stage16(gsrc + knx, ldst + ((s + 1) & 1) * BTILE);
    WAIT_ASYNC(1);                              // tile s has landed
    __syncthreads();

    v16bf a = load_fragA(X + (size_t)m0 * H_ + k0, H_);
    if (k0 + 64 < H_)
      __builtin_prefetch(X + (size_t)m0 * H_ + k0 + 64, 0, 3);
    const bf16* bb = &bstage[s & 1][0];
    for (int t = 0; t < 4; t++)
      c[t] = wmma_bf16(a, load_fragB(bb + (size_t)(t * 16) * BPITCH, BPITCH), c[t]);
    __syncthreads();                            // buffer may be overwritten next
  }

  for (int t = 0; t < 4; t++) {
    int oc = n0 + t * 16 + n;
    int head = oc >> 6, d = oc & 63;
    for (int r = 0; r < 8; r++) {
      int s = m0 + r + 8 * hf;                  // C layout: m = vgpr + 8*(lane/16)
      bf16 val = (bf16)c[t][r];
      if (z == 0)      Q [((size_t)head * S_ + s) * HD_ + d] = val;
      else if (z == 1) K [((size_t)head * S_ + s) * HD_ + d] = val;
      else             Vt[((size_t)head * HD_ + d) * S_ + s] = val;
    }
  }
}

// ---- block-sparse flash attention ------------------------------------------
// grid = (64 query blocks, 16 heads), 128 threads (4 waves, 16 rows each).
// attention_mask is a bool[S][S] expanded block-uniformly -> test 1 byte/block.
__global__ void __launch_bounds__(128)
sparse_attn(const bf16* __restrict__ Q, const bf16* __restrict__ K,
            const bf16* __restrict__ Vt, const unsigned char* __restrict__ mask,
            bf16* __restrict__ attn) {
  __shared__ bf16 pbuf[4 * 16 * 64];           // per-wave 16x64 P staging, 8 KB
  const int qb = blockIdx.x, head = blockIdx.y;
  const int wave = threadIdx.x >> 5, lane = threadIdx.x & 31;
  const int n = lane & 15, hf = lane >> 4;
  const bf16* Qh = Q  + (size_t)head * S_ * HD_;
  const bf16* Kh = K  + (size_t)head * S_ * HD_;
  const bf16* Vh = Vt + (size_t)head * HD_ * S_;
  const int q0 = qb * BS_ + wave * 16;
  const float scale = 0.125f;                  // HD^-0.5

  v16bf aq0 = load_fragA(Qh + (size_t)q0 * HD_ + 0,  HD_);
  v16bf aq1 = load_fragA(Qh + (size_t)q0 * HD_ + 32, HD_);

  v8f o[4];
  float mrow[8], lrow[8];
  for (int t = 0; t < 4; t++)
    for (int r = 0; r < 8; r++) o[t][r] = 0.f;
  for (int r = 0; r < 8; r++) { mrow[r] = -1e30f; lrow[r] = 0.f; }

  bf16* myp = pbuf + wave * 16 * 64;

  for (int kb = 0; kb < NB_; kb++) {
    // one byte decides the whole 64x64 block (mask is block-uniform)
    if (!mask[(size_t)(qb * BS_) * S_ + (size_t)kb * BS_]) continue;

    // S = Q Kt, 16x64 strip per wave (8 WMMAs)
    v8f sc[4];
    for (int t = 0; t < 4; t++) {
      v8f acc;
      for (int r = 0; r < 8; r++) acc[r] = 0.f;
      acc = wmma_bf16(aq0, load_fragB(Kh + (size_t)(kb * BS_ + t * 16) * HD_ + 0,  HD_), acc);
      acc = wmma_bf16(aq1, load_fragB(Kh + (size_t)(kb * BS_ + t * 16) * HD_ + 32, HD_), acc);
      for (int r = 0; r < 8; r++) acc[r] *= scale;
      sc[t] = acc;
    }

    // row max across the 4 n-tiles, then across the 16-lane group
    float tm[8];
    for (int r = 0; r < 8; r++)
      tm[r] = fmaxf(fmaxf(sc[0][r], sc[1][r]), fmaxf(sc[2][r], sc[3][r]));
    for (int off = 1; off < 16; off <<= 1)
      for (int r = 0; r < 8; r++)
        tm[r] = fmaxf(tm[r], __shfl_xor(tm[r], off, 32));

    float mnew[8], corr[8];
    for (int r = 0; r < 8; r++) {
      mnew[r] = fmaxf(mrow[r], tm[r]);
      corr[r] = __expf(mrow[r] - mnew[r]);
      mrow[r] = mnew[r];
    }

    __syncthreads();   // uniform across block: all waves share qb/mask row
    float psum[8];
    for (int r = 0; r < 8; r++) psum[r] = 0.f;
    for (int t = 0; t < 4; t++)
      for (int r = 0; r < 8; r++) {
        float p = __expf(sc[t][r] - mnew[r]);
        psum[r] += p;
        myp[(r + 8 * hf) * 64 + t * 16 + n] = (bf16)p;   // ds_store, C->row-major
      }
    __syncthreads();

    for (int off = 1; off < 16; off <<= 1)
      for (int r = 0; r < 8; r++) psum[r] += __shfl_xor(psum[r], off, 32);
    for (int r = 0; r < 8; r++) lrow[r] = lrow[r] * corr[r] + psum[r];

    // O = O*corr + P V  (A-fragments of P come back from LDS via ds_load)
    v16bf p0 = load_fragA(myp + 0,  64);
    v16bf p1 = load_fragA(myp + 32, 64);
    for (int t = 0; t < 4; t++) {
      v8f oc = o[t];
      for (int r = 0; r < 8; r++) oc[r] *= corr[r];
      oc = wmma_bf16(p0, load_fragB(Vh + (size_t)(t * 16) * S_ + kb * BS_ + 0,  S_), oc);
      oc = wmma_bf16(p1, load_fragB(Vh + (size_t)(t * 16) * S_ + kb * BS_ + 32, S_), oc);
      o[t] = oc;
    }
  }

  // normalize and emit attn [s][head*64+d] as bf16 for the output GEMM
  for (int t = 0; t < 4; t++) {
    int d = t * 16 + n;
    for (int r = 0; r < 8; r++) {
      int s = q0 + r + 8 * hf;
      attn[(size_t)s * H_ + head * HD_ + d] = (bf16)(o[t][r] / lrow[r]);
    }
  }
}

// ---- output projection: out = attn @ Wo^T + bo (fp32 out) ------------------
__global__ void __launch_bounds__(256)
out_gemm(const bf16* __restrict__ A, const bf16* __restrict__ W,
         const float* __restrict__ bias, float* __restrict__ out) {
  __shared__ bf16 bstage[2][BTILE];
  const int wave = threadIdx.x >> 5, lane = threadIdx.x & 31;
  const int n0 = blockIdx.x * 64;
  const int m0 = blockIdx.y * 128 + wave * 16;
  const int n = lane & 15, hf = lane >> 4;

  const int crow = threadIdx.x >> 2, cchunk = threadIdx.x & 3;
  const bf16* gsrc = W + (size_t)(n0 + crow) * H_ + cchunk * 8;
  bf16* ldst = &bstage[0][crow * BPITCH + cchunk * 8];

  v8f c[4];
  for (int t = 0; t < 4; t++) {
    float b = bias[n0 + t * 16 + n];
    for (int r = 0; r < 8; r++) c[t][r] = b;
  }

  stage16(gsrc, ldst);
  for (int s = 0; s < H_ / 32; s++) {
    const int k0 = s * 32;
    const int knx = (s + 1 < H_ / 32) ? (k0 + 32) : k0;
    stage16(gsrc + knx, ldst + ((s + 1) & 1) * BTILE);
    WAIT_ASYNC(1);
    __syncthreads();

    v16bf a = load_fragA(A + (size_t)m0 * H_ + k0, H_);
    if (k0 + 64 < H_)
      __builtin_prefetch(A + (size_t)m0 * H_ + k0 + 64, 0, 3);
    const bf16* bb = &bstage[s & 1][0];
    for (int t = 0; t < 4; t++)
      c[t] = wmma_bf16(a, load_fragB(bb + (size_t)(t * 16) * BPITCH, BPITCH), c[t]);
    __syncthreads();
  }

  for (int t = 0; t < 4; t++)
    for (int r = 0; r < 8; r++)
      out[(size_t)(m0 + r + 8 * hf) * H_ + n0 + t * 16 + n] = c[t][r];
}

// ---------------------------------------------------------------------------
extern "C" void kernel_launch(void* const* d_in, const int* in_sizes, int n_in,
                              void* d_out, int out_size, void* d_ws, size_t ws_size,
                              hipStream_t stream) {
  (void)in_sizes; (void)n_in; (void)out_size; (void)ws_size;
  const float* x  = (const float*)d_in[0];
  const float* wq = (const float*)d_in[1];
  const float* bq = (const float*)d_in[2];
  const float* wk = (const float*)d_in[3];
  const float* bk = (const float*)d_in[4];
  const float* wv = (const float*)d_in[5];
  const float* bv = (const float*)d_in[6];
  const float* wo = (const float*)d_in[7];
  const float* bo = (const float*)d_in[8];
  // JAX bool array: 1 byte per element
  const unsigned char* mask = (const unsigned char*)d_in[9];

  char* ws = (char*)d_ws;
  size_t off = 0;
  bf16* Xb  = (bf16*)(ws + off); off += (size_t)S_ * H_ * 2;   //  8 MB
  bf16* Wqb = (bf16*)(ws + off); off += (size_t)H_ * H_ * 2;   //  2 MB
  bf16* Wkb = (bf16*)(ws + off); off += (size_t)H_ * H_ * 2;
  bf16* Wvb = (bf16*)(ws + off); off += (size_t)H_ * H_ * 2;
  bf16* Wob = (bf16*)(ws + off); off += (size_t)H_ * H_ * 2;
  bf16* Qb  = (bf16*)(ws + off); off += (size_t)S_ * H_ * 2;   //  8 MB
  bf16* Kb  = (bf16*)(ws + off); off += (size_t)S_ * H_ * 2;
  bf16* Vtb = (bf16*)(ws + off); off += (size_t)S_ * H_ * 2;
  bf16* At  = (bf16*)(ws + off); off += (size_t)S_ * H_ * 2;   // attn output

  // 1) convert to bf16
  cvt_f32_bf16<<<4096, 256, 0, stream>>>(x,  Xb,  S_ * H_);
  cvt_f32_bf16<<<2048, 256, 0, stream>>>(wq, Wqb, H_ * H_);
  cvt_f32_bf16<<<2048, 256, 0, stream>>>(wk, Wkb, H_ * H_);
  cvt_f32_bf16<<<2048, 256, 0, stream>>>(wv, Wvb, H_ * H_);
  cvt_f32_bf16<<<2048, 256, 0, stream>>>(wo, Wob, H_ * H_);

  // 2) fused QKV projection (z picks Q/K/V)
  qkv_gemm<<<dim3(H_ / 64, S_ / 128, 3), 256, 0, stream>>>(
      Xb, Wqb, Wkb, Wvb, bq, bk, bv, Qb, Kb, Vtb);

  // 3) block-sparse flash attention
  sparse_attn<<<dim3(NB_, NH_), 128, 0, stream>>>(Qb, Kb, Vtb, mask, At);

  // 4) output projection -> fp32 d_out
  out_gemm<<<dim3(H_ / 64, S_ / 128), 256, 0, stream>>>(
      At, Wob, bo, (float*)d_out);
}
